// EinsumSelfAttention_38259568673100
// MI455X (gfx1250) — compile-verified, matrix operand order checked by reading, not run on hardware
//
#include <hip/hip_runtime.h>

#define EMB   1024
#define HEADS 16
#define DK    64
#define SEQ   2048
#define BATCH 2
#define MROWS (BATCH * SEQ) /* 4096 */

typedef __bf16 bf16;
typedef __attribute__((ext_vector_type(16))) __bf16 v16bf;
typedef __attribute__((ext_vector_type(8)))  __bf16 v8bf;
typedef __attribute__((ext_vector_type(8)))  float  v8f;

#define WMMA_BF16(a, b, c) \
  __builtin_amdgcn_wmma_f32_16x16x32_bf16(false, (a), false, (b), (short)0, (c), false, false)

// ---------------------------------------------------------------------------
// Fragment load: 16-bit A-layout (and col-major B via row loads of W / K / Vt)
// lane L (L<16):  row = row0+L,    halves 0..7 = K[k0..k0+7],   8..15 = K[k0+16..k0+23]
// lane L (L>=16): row = row0+L-16, halves 0..7 = K[k0+8..k0+15],8..15 = K[k0+24..k0+31]
// => two contiguous 16B loads per lane (global_load_b128 x2).
// ---------------------------------------------------------------------------
static __device__ __forceinline__ v16bf load_frag_g(const bf16* __restrict__ base,
                                                    int ld, int row0, int k0) {
  const int lane = threadIdx.x & 31;
  const bf16* p = base + (size_t)(row0 + (lane & 15)) * ld + k0 + ((lane >> 4) << 3);
  v8bf lo = *(const v8bf*)p;
  v8bf hi = *(const v8bf*)(p + 16);
  v16bf r;
#pragma unroll
  for (int i = 0; i < 8; ++i) { r[i] = lo[i]; r[i + 8] = hi[i]; }
  return r;
}

static __device__ __forceinline__ v16bf load_frag_lds(const bf16* base /*16x32 row-major*/) {
  const int lane = threadIdx.x & 31;
  const bf16* p = base + (lane & 15) * 32 + ((lane >> 4) << 3);
  v8bf lo = *(const v8bf*)p;
  v8bf hi = *(const v8bf*)(p + 16);
  v16bf r;
#pragma unroll
  for (int i = 0; i < 8; ++i) { r[i] = lo[i]; r[i + 8] = hi[i]; }
  return r;
}

// ---------------------------------------------------------------------------
// fp32 -> bf16 conversion
// ---------------------------------------------------------------------------
__global__ void cvt_bf16_kernel(const float* __restrict__ in, bf16* __restrict__ out, int n) {
  int i = blockIdx.x * 256 + threadIdx.x;
  if (i < n) out[i] = (bf16)in[i];
}

// ---------------------------------------------------------------------------
// GEMM: C[M,N] = A[M,K] * W[N,K]^T + bias[N]
// MODE 0: fp32 row-major out; MODE 1: bf16 row-major out;
// MODE 2: bf16 out transposed per-head: Vt[((b*H+h)*DK+d)*SEQ + t]
// Block = 128 threads (4 waves, 2Mx2N); wave computes 64x32 via 4x2 WMMA tiles
// (8 WMMAs per 6 fragment loads).
// ---------------------------------------------------------------------------
template <int MODE>
__global__ __launch_bounds__(128) void gemm_wmma_kernel(
    const bf16* __restrict__ A, const bf16* __restrict__ W,
    const float* __restrict__ bias, float* __restrict__ outF,
    bf16* __restrict__ outB, int Ndim, int Kdim) {
  const int w    = threadIdx.x >> 5;
  const int lane = threadIdx.x & 31;
  const int m0 = blockIdx.y * 128 + (w & 1) * 64;
  const int n0 = blockIdx.x * 64 + (w >> 1) * 32;

  v8f acc[4][2] = {};
  for (int k0 = 0; k0 < Kdim; k0 += 32) {
    // speculative prefetch two k-steps ahead (global_prefetch_b8 path)
    if (k0 + 64 < Kdim) {
      __builtin_prefetch(A + (size_t)(m0 + (lane & 63)) * Kdim + k0 + 64, 0, 1);
      __builtin_prefetch(W + (size_t)(n0 + (lane & 31)) * Kdim + k0 + 64, 0, 1);
    }
    v16bf af[4], bf[2];
#pragma unroll
    for (int i = 0; i < 4; ++i) af[i] = load_frag_g(A, Kdim, m0 + 16 * i, k0);
#pragma unroll
    for (int j = 0; j < 2; ++j) bf[j] = load_frag_g(W, Kdim, n0 + 16 * j, k0);
#pragma unroll
    for (int i = 0; i < 4; ++i)
#pragma unroll
      for (int j = 0; j < 2; ++j)
        acc[i][j] = WMMA_BF16(af[i], bf[j], acc[i][j]);
  }

  const int rbase = (lane >> 4) << 3; // +8 rows for upper half-wave
  const int cb    = lane & 15;
#pragma unroll
  for (int i = 0; i < 4; ++i) {
#pragma unroll
    for (int j = 0; j < 2; ++j) {
#pragma unroll
      for (int r = 0; r < 8; ++r) {
        int row = m0 + 16 * i + r + rbase;
        int col = n0 + 16 * j + cb;
        float v = acc[i][j][r] + bias[col];
        if (MODE == 0) {
          outF[(size_t)row * Ndim + col] = v;
        } else if (MODE == 1) {
          outB[(size_t)row * Ndim + col] = (bf16)v;
        } else {
          int t = row & (SEQ - 1), b = row >> 11;
          int h = col >> 6, d = col & 63;
          outB[(((size_t)(b * HEADS + h)) * DK + d) * SEQ + t] = (bf16)v;
        }
      }
    }
  }
}

// ---------------------------------------------------------------------------
// Flash attention: one wave per (b, h, 16-query tile). 32-key chunks.
// Q,K: [B*T, EMB] bf16 (head offset h*DK).  Vt: [B*H, DK, SEQ] bf16.
// Output: [B*T, EMB] bf16 (already softmax-normalized).
// ---------------------------------------------------------------------------
__global__ __launch_bounds__(128) void attn_wmma_kernel(
    const bf16* __restrict__ Q, const bf16* __restrict__ K,
    const bf16* __restrict__ Vt, bf16* __restrict__ O) {
  __shared__ __align__(16) bf16 plds[4][16 * 32];

  const int w    = threadIdx.x >> 5;
  const int lane = threadIdx.x & 31;
  const int gw = blockIdx.x * 4 + w;
  const int tq = gw & 127;          // query tile within sequence
  const int h  = (gw >> 7) & 15;    // head
  const int b  = gw >> 11;          // batch

  const bf16* Qbase = Q + (size_t)b * SEQ * EMB + h * DK;
  const bf16* Kbase = K + (size_t)b * SEQ * EMB + h * DK;
  const bf16* Vbase = Vt + ((size_t)(b * HEADS + h)) * DK * SEQ;
  bf16* myp = &plds[w][0];

  // Q fragments are invariant across the key loop (d_k = 64 = 2 K-steps)
  v16bf qf0 = load_frag_g(Qbase, EMB, tq * 16, 0);
  v16bf qf1 = load_frag_g(Qbase, EMB, tq * 16, 32);

  v8f o0 = {}, o1 = {}, o2 = {}, o3 = {};
  float mrun[8], lrun[8];
#pragma unroll
  for (int r = 0; r < 8; ++r) { mrun[r] = -1e30f; lrun[r] = 0.0f; }

  const float scale = 0.125f; // 1/sqrt(64)
  const int rbase = (lane >> 4) << 3;

  for (int s0 = 0; s0 < SEQ; s0 += 32) {
    // prefetch next key/value chunk while this one computes
    if (s0 + 32 < SEQ) {
      __builtin_prefetch(Kbase + (size_t)(s0 + 32 + lane) * EMB, 0, 1);
      __builtin_prefetch(Vbase + (size_t)(lane * 2) * SEQ + s0 + 32, 0, 1);
    }
    // S tile (16 q-rows x 32 key-cols) = two 16x16 WMMA accumulators
    v16bf kf00 = load_frag_g(Kbase, EMB, s0, 0);
    v16bf kf01 = load_frag_g(Kbase, EMB, s0, 32);
    v16bf kf10 = load_frag_g(Kbase, EMB, s0 + 16, 0);
    v16bf kf11 = load_frag_g(Kbase, EMB, s0 + 16, 32);
    v8f sa = {}, sb = {};
    sa = WMMA_BF16(qf0, kf00, sa);
    sa = WMMA_BF16(qf1, kf01, sa);
    sb = WMMA_BF16(qf0, kf10, sb);
    sb = WMMA_BF16(qf1, kf11, sb);

    // Online softmax over the 32 columns; row stats per 16-lane group
#pragma unroll
    for (int r = 0; r < 8; ++r) {
      float x0 = sa[r] * scale, x1 = sb[r] * scale;
      float mx = fmaxf(x0, x1);
#pragma unroll
      for (int off = 1; off < 16; off <<= 1) mx = fmaxf(mx, __shfl_xor(mx, off, 16));
      float mnew = fmaxf(mrun[r], mx);
      float corr = __expf(mrun[r] - mnew);
      float p0 = __expf(x0 - mnew);
      float p1 = __expf(x1 - mnew);
      float rs = p0 + p1;
#pragma unroll
      for (int off = 1; off < 16; off <<= 1) rs += __shfl_xor(rs, off, 16);
      lrun[r] = lrun[r] * corr + rs;
      mrun[r] = mnew;
      o0[r] *= corr; o1[r] *= corr; o2[r] *= corr; o3[r] *= corr;
      // stage P (bf16) to per-wave LDS tile (16x32, row-major)
      int prow = r + rbase;
      myp[prow * 32 + (lane & 15)]      = (bf16)p0;
      myp[prow * 32 + 16 + (lane & 15)] = (bf16)p1;
    }

    // re-read P as an A-fragment (intra-wave LDS transpose), V col-major via Vt rows
    v16bf pf  = load_frag_lds(myp);
    v16bf vf0 = load_frag_g(Vbase, SEQ, 0,  s0);
    v16bf vf1 = load_frag_g(Vbase, SEQ, 16, s0);
    v16bf vf2 = load_frag_g(Vbase, SEQ, 32, s0);
    v16bf vf3 = load_frag_g(Vbase, SEQ, 48, s0);
    o0 = WMMA_BF16(pf, vf0, o0);
    o1 = WMMA_BF16(pf, vf1, o1);
    o2 = WMMA_BF16(pf, vf2, o2);
    o3 = WMMA_BF16(pf, vf3, o3);
  }

  // Normalize and write per-head output slice as bf16
#pragma unroll
  for (int r = 0; r < 8; ++r) {
    float inv = 1.0f / lrun[r];
    int trow = tq * 16 + r + rbase;
    size_t off = ((size_t)b * SEQ + trow) * EMB + h * DK + (lane & 15);
    O[off + 0]  = (bf16)(o0[r] * inv);
    O[off + 16] = (bf16)(o1[r] * inv);
    O[off + 32] = (bf16)(o2[r] * inv);
    O[off + 48] = (bf16)(o3[r] * inv);
  }
}

// ---------------------------------------------------------------------------
// Residual + LayerNorm: out = LN(x + proj) * gamma + beta; one block per row.
// ---------------------------------------------------------------------------
__global__ __launch_bounds__(256) void ln_kernel(const float* __restrict__ x,
                                                 const float* __restrict__ proj,
                                                 const float* __restrict__ gamma,
                                                 const float* __restrict__ beta,
                                                 float* __restrict__ out) {
  __shared__ float rs[256], rs2[256];
  const int row = blockIdx.x;
  const float* xr = x + (size_t)row * EMB;
  const float* pr = proj + (size_t)row * EMB;
  float vals[4];
  float s = 0.0f, s2 = 0.0f;
#pragma unroll
  for (int i = 0; i < 4; ++i) {
    int idx = threadIdx.x + i * 256;
    float v = xr[idx] + pr[idx];
    vals[i] = v;
    s += v;
    s2 += v * v;
  }
  rs[threadIdx.x] = s;
  rs2[threadIdx.x] = s2;
  __syncthreads();
  for (int o = 128; o > 0; o >>= 1) {
    if (threadIdx.x < o) {
      rs[threadIdx.x] += rs[threadIdx.x + o];
      rs2[threadIdx.x] += rs2[threadIdx.x + o];
    }
    __syncthreads();
  }
  float mu  = rs[0] * (1.0f / EMB);
  float var = rs2[0] * (1.0f / EMB) - mu * mu;
  float inv = rsqrtf(var + 1e-5f);
#pragma unroll
  for (int i = 0; i < 4; ++i) {
    int idx = threadIdx.x + i * 256;
    out[(size_t)row * EMB + idx] = (vals[i] - mu) * inv * gamma[idx] + beta[idx];
  }
}

// ---------------------------------------------------------------------------
extern "C" void kernel_launch(void* const* d_in, const int* in_sizes, int n_in,
                              void* d_out, int out_size, void* d_ws, size_t ws_size,
                              hipStream_t stream) {
  (void)in_sizes; (void)n_in; (void)out_size; (void)ws_size;
  const float* x     = (const float*)d_in[0];
  const float* wq    = (const float*)d_in[1];
  const float* bq    = (const float*)d_in[2];
  const float* wk    = (const float*)d_in[3];
  const float* bk    = (const float*)d_in[4];
  const float* wv    = (const float*)d_in[5];
  const float* bv    = (const float*)d_in[6];
  const float* wo    = (const float*)d_in[7];
  const float* bo    = (const float*)d_in[8];
  const float* gamma = (const float*)d_in[9];
  const float* beta  = (const float*)d_in[10];
  float* out = (float*)d_out;

  // Workspace carve-up (~64 MB total)
  char* p = (char*)d_ws;
  bf16* xb    = (bf16*)p; p += (size_t)MROWS * EMB * sizeof(bf16);
  bf16* wqb   = (bf16*)p; p += (size_t)EMB * EMB * sizeof(bf16);
  bf16* wkb   = (bf16*)p; p += (size_t)EMB * EMB * sizeof(bf16);
  bf16* wvb   = (bf16*)p; p += (size_t)EMB * EMB * sizeof(bf16);
  bf16* wob   = (bf16*)p; p += (size_t)EMB * EMB * sizeof(bf16);
  bf16* Qb    = (bf16*)p; p += (size_t)MROWS * EMB * sizeof(bf16);
  bf16* Kb    = (bf16*)p; p += (size_t)MROWS * EMB * sizeof(bf16);
  bf16* Vtb   = (bf16*)p; p += (size_t)MROWS * EMB * sizeof(bf16);
  bf16* attnb = (bf16*)p; p += (size_t)MROWS * EMB * sizeof(bf16);
  float* projf = (float*)p;

  // 1) fp32 -> bf16 conversions
  int nx = MROWS * EMB;
  cvt_bf16_kernel<<<nx / 256, 256, 0, stream>>>(x, xb, nx);
  int nw = EMB * EMB;
  cvt_bf16_kernel<<<nw / 256, 256, 0, stream>>>(wq, wqb, nw);
  cvt_bf16_kernel<<<nw / 256, 256, 0, stream>>>(wk, wkb, nw);
  cvt_bf16_kernel<<<nw / 256, 256, 0, stream>>>(wv, wvb, nw);
  cvt_bf16_kernel<<<nw / 256, 256, 0, stream>>>(wo, wob, nw);

  // 2) Q/K/V projections (V written per-head transposed)
  dim3 gg(EMB / 64, MROWS / 128);
  gemm_wmma_kernel<1><<<gg, 128, 0, stream>>>(xb, wqb, bq, nullptr, Qb, EMB, EMB);
  gemm_wmma_kernel<1><<<gg, 128, 0, stream>>>(xb, wkb, bk, nullptr, Kb, EMB, EMB);
  gemm_wmma_kernel<2><<<gg, 128, 0, stream>>>(xb, wvb, bv, nullptr, Vtb, EMB, EMB);

  // 3) Flash attention: B*H*(T/16) = 4096 waves, 4 waves/block
  attn_wmma_kernel<<<(BATCH * HEADS * (SEQ / 16)) / 4, 128, 0, stream>>>(Qb, Kb, Vtb, attnb);

  // 4) Output projection (fp32 result for LayerNorm)
  gemm_wmma_kernel<0><<<gg, 128, 0, stream>>>(attnb, wob, bo, projf, nullptr, EMB, EMB);

  // 5) Residual + LayerNorm
  ln_kernel<<<MROWS, 256, 0, stream>>>(x, projf, gamma, beta, out);
}
